// CAPMemory_part_44607530336552
// MI455X (gfx1250) — compile-verified, matrix operand order checked by reading, not run on hardware
//
#include <hip/hip_runtime.h>
#include <hip/hip_bf16.h>

// ---------------------------------------------------------------------------
// Problem constants (from reference): N=512, n_cams=8, M=2048, D=4096
// P = n_cams*M = 16384, K per GEMM = 2048, BETA = 0.05 (1/BETA = 20)
// ---------------------------------------------------------------------------
#define NROWS   512
#define NCAMS   8
#define MMEM    2048
#define DDIM    4096
#define PCOLS   16384
#define KHALF   2048
#define INVB    20.0f
#define NEGINF  (-3.0e38f)

typedef __attribute__((ext_vector_type(16))) __bf16 v16bf;
typedef __attribute__((ext_vector_type(8)))  __bf16 v8bf;
typedef __attribute__((ext_vector_type(4)))  __bf16 v4bf;
typedef __attribute__((ext_vector_type(8)))  float  v8f;

// ---------------------------------------------------------------------------
// Kernel 0: zero the scalar output
// ---------------------------------------------------------------------------
__global__ void zero_out_kernel(float* out) { out[0] = 0.0f; }

// ---------------------------------------------------------------------------
// Kernel 1: convert features f32 -> bf16 (512x4096, 4 elems/thread)
// ---------------------------------------------------------------------------
__global__ __launch_bounds__(256) void cvt_feat_kernel(const float* __restrict__ f,
                                                       __bf16* __restrict__ o) {
    int i = (blockIdx.x * 256 + threadIdx.x) * 4;
    float4 v = *(const float4*)(f + i);
    v4bf b;
    b[0] = (__bf16)v.x; b[1] = (__bf16)v.y; b[2] = (__bf16)v.z; b[3] = (__bf16)v.w;
    *(v4bf*)(o + i) = b;
}

// ---------------------------------------------------------------------------
// Kernel 2: GEMM  sims_g[n,p] = sum_k fbf[n, g*2048+k] * mem[p, g*2048+k]
//   block: 256 threads (8 wave32). Block tile = 512 rows x 64 cols.
//   wave tile = 64 rows x 64 cols = 4x4 WMMA 16x16 tiles, K step 32 (bf16).
//   B (memory) is f32 in HBM, converted to bf16 on the fly (read once).
// ---------------------------------------------------------------------------
__global__ __launch_bounds__(256) void gemm_bf16_kernel(const __bf16* __restrict__ fbf,
                                                        const float* __restrict__ mem,
                                                        float* __restrict__ sims,
                                                        int g) {
    const int lane = threadIdx.x & 31;
    const int wave = threadIdx.x >> 5;
    const int half = lane >> 4;      // 0: lanes 0-15, 1: lanes 16-31
    const int lr   = lane & 15;
    const int colBase = blockIdx.x * 64;
    const int rowBase = wave * 64;
    const int kOff    = g * KHALF;

    v8f acc[4][4];
#pragma unroll
    for (int rt = 0; rt < 4; ++rt)
#pragma unroll
        for (int ct = 0; ct < 4; ++ct)
#pragma unroll
            for (int e = 0; e < 8; ++e) acc[rt][ct][e] = 0.0f;

    for (int k = 0; k < KHALF; k += 32) {
        // A fragments: lane lr holds row M=lr; K chunks {k+half*8 .. +8} and {k+16+half*8 .. +8}
        v16bf afrag[4];
#pragma unroll
        for (int rt = 0; rt < 4; ++rt) {
            const __bf16* ap = fbf + (size_t)(rowBase + rt * 16 + lr) * DDIM + kOff + k + half * 8;
            v8bf lo = *(const v8bf*)ap;
            v8bf hi = *(const v8bf*)(ap + 16);
            afrag[rt] = __builtin_shufflevector(lo, hi, 0,1,2,3,4,5,6,7,8,9,10,11,12,13,14,15);
        }
#pragma unroll
        for (int ct = 0; ct < 4; ++ct) {
            const float* bp = mem + (size_t)(colBase + ct * 16 + lr) * DDIM + kOff + k + half * 8;
            v8f b0 = *(const v8f*)bp;
            v8f b1 = *(const v8f*)(bp + 16);
            v16bf bfrag;
#pragma unroll
            for (int e = 0; e < 8; ++e) {
                bfrag[e]     = (__bf16)b0[e];
                bfrag[e + 8] = (__bf16)b1[e];
            }
#pragma unroll
            for (int rt = 0; rt < 4; ++rt) {
#ifdef __HIP_DEVICE_COMPILE__
                acc[rt][ct] = __builtin_amdgcn_wmma_f32_16x16x32_bf16(
                    /*neg_a=*/false, afrag[rt], /*neg_b=*/false, bfrag,
                    /*c_mod=*/(short)0, acc[rt][ct], /*reuse_a=*/false, /*reuse_b=*/false);
#endif
            }
        }
    }

    // C/D layout: VGPR v, lanes 0-15 -> M=v, lanes 16-31 -> M=v+8; N = lane&15
#pragma unroll
    for (int rt = 0; rt < 4; ++rt)
#pragma unroll
        for (int ct = 0; ct < 4; ++ct) {
            int col = colBase + ct * 16 + lr;
#pragma unroll
            for (int v = 0; v < 8; ++v) {
                int row = rowBase + rt * 16 + half * 8 + v;
                sims[(size_t)row * PCOLS + col] = acc[rt][ct][v];
            }
        }
}

// ---------------------------------------------------------------------------
// Block-wide argmax over rowbuf[base .. base+len) -> bcf[slot], bci[slot]
// ---------------------------------------------------------------------------
__device__ __forceinline__ void block_argmax(const float* rowbuf, int base, int len,
                                             float* redv, int* redi,
                                             float* bcf, int* bci, int slot, int tid) {
    float best = NEGINF;
    int   bi   = base;
    for (int i = tid; i < len; i += 256) {
        float v = rowbuf[base + i];
        if (v > best) { best = v; bi = base + i; }
    }
    redv[tid] = best; redi[tid] = bi;
    __syncthreads();
    if (tid == 0) {
        float m = redv[0]; int mi = redi[0];
        for (int t = 1; t < 256; ++t)
            if (redv[t] > m) { m = redv[t]; mi = redi[t]; }
        bcf[slot] = m; bci[slot] = mi;
    }
    __syncthreads();
}

// ---------------------------------------------------------------------------
// Kernel 3: per-row losses. One block (256 thr) per row; row staged in LDS.
//   ce:     logsumexp over the 2048-wide cam block minus proxy logit
//   assoc:  top-50 (proxy masked) logsumexp w/ proxy logit minus proxy logit
//   online: per-cam maxima -> top-3 cams, mask them, top-30 of rest,
//           logsumexp over 33 minus mean of 3 cam-top logits
// ---------------------------------------------------------------------------
__global__ __launch_bounds__(256) void loss_kernel(const float* __restrict__ sims0,
                                                   const float* __restrict__ sims1,
                                                   const int* __restrict__ cams,
                                                   const int* __restrict__ proxy,
                                                   float* __restrict__ out) {
    extern __shared__ char smem[];
    float* rowbuf = (float*)smem;            // 16384 f32
    float* redv   = rowbuf + PCOLS;          // 256
    int*   redi   = (int*)(redv + 256);      // 256
    float* sval   = (float*)(redi + 256);    // 64
    int*   sidx   = (int*)(sval + 64);       // 64
    float* bcf    = (float*)(sidx + 64);     // 32 broadcast scalars
    int*   bci    = (int*)(bcf + 32);        // 32 broadcast indices

    const int tid  = threadIdx.x;
    const int n    = blockIdx.x;
    const int cam  = cams[n];
    const int prox = proxy[n];               // global index in [0, 16384)

    float total = 0.0f;                      // meaningful on thread 0 only

    for (int gidx = 0; gidx < 2; ++gidx) {
        const float* row = (gidx == 0 ? sims0 : sims1) + (size_t)n * PCOLS;
        for (int i = tid; i < PCOLS; i += 256) rowbuf[i] = row[i];
        __syncthreads();

        const float posval = rowbuf[prox];

        // ---- ce: logsumexp over cam block ----
        block_argmax(rowbuf, cam * MMEM, MMEM, redv, redi, bcf, bci, 0, tid);
        const float maxb = bcf[0];
        {
            float s = 0.0f;
            for (int i = tid; i < MMEM; i += 256)
                s += __expf((rowbuf[cam * MMEM + i] - maxb) * INVB);
            redv[tid] = s;
            __syncthreads();
            if (tid == 0) {
                float acc = 0.0f;
                for (int t = 0; t < 256; ++t) acc += redv[t];
                bcf[15] = acc;
            }
            __syncthreads();
        }
        const float ce = (maxb - posval) * INVB + __logf(bcf[15]);

        // ---- per-cam maxima (slots 4..11) ----
        for (int c = 0; c < NCAMS; ++c)
            block_argmax(rowbuf, c * MMEM, MMEM, redv, redi, bcf, bci, 4 + c, tid);

        // ---- top-3 cams (thread 0) -> slots 1..3 ----
        if (tid == 0) {
            float cv[NCAMS]; int ci[NCAMS];
            for (int c = 0; c < NCAMS; ++c) { cv[c] = bcf[4 + c]; ci[c] = bci[4 + c]; }
            for (int j = 0; j < 3; ++j) {
                int best = j;
                for (int c = j + 1; c < NCAMS; ++c)
                    if (cv[c] > cv[best]) best = c;
                float tv = cv[j]; cv[j] = cv[best]; cv[best] = tv;
                int   ti = ci[j]; ci[j] = ci[best]; ci[best] = ti;
                bcf[1 + j] = cv[j]; bci[1 + j] = ci[j];
            }
        }
        __syncthreads();

        // ---- assoc: mask proxy, extract top-50 ----
        if (tid == 0) rowbuf[prox] = NEGINF;
        __syncthreads();
        for (int it = 0; it < 50; ++it) {
            block_argmax(rowbuf, 0, PCOLS, redv, redi, bcf, bci, 14, tid);
            if (tid == 0) {
                sval[it] = bcf[14]; sidx[it] = bci[14];
                rowbuf[bci[14]] = NEGINF;
            }
            __syncthreads();
        }
        if (tid == 0) {
            float m = sval[0] > posval ? sval[0] : posval;
            float s = __expf((posval - m) * INVB);
            for (int i = 0; i < 50; ++i) s += __expf((sval[i] - m) * INVB);
            bcf[12] = (m - posval) * INVB + __logf(s);
            // restore the row, then mask the 3 cam-top positions for "online"
            rowbuf[prox] = posval;
            for (int i = 0; i < 50; ++i) rowbuf[sidx[i]] = sval[i];
            rowbuf[bci[1]] = NEGINF;
            rowbuf[bci[2]] = NEGINF;
            rowbuf[bci[3]] = NEGINF;
        }
        __syncthreads();

        // ---- online: top-30 of remaining ----
        for (int it = 0; it < 30; ++it) {
            block_argmax(rowbuf, 0, PCOLS, redv, redi, bcf, bci, 14, tid);
            if (tid == 0) {
                sval[it] = bcf[14];
                rowbuf[bci[14]] = NEGINF;
            }
            __syncthreads();
        }
        if (tid == 0) {
            const float p1 = bcf[1], p2 = bcf[2], p3 = bcf[3];
            float m = p1;
            if (p2 > m) m = p2;
            if (p3 > m) m = p3;
            if (sval[0] > m) m = sval[0];
            float s = __expf((p1 - m) * INVB) + __expf((p2 - m) * INVB) + __expf((p3 - m) * INVB);
            for (int i = 0; i < 30; ++i) s += __expf((sval[i] - m) * INVB);
            bcf[13] = m * INVB + __logf(s) - (p1 + p2 + p3) * INVB * (1.0f / 3.0f);
        }
        __syncthreads();

        if (tid == 0)
            total += 0.6f * ce + 0.7f * bcf[12] + 0.7f * bcf[13];
        __syncthreads();   // before rowbuf reload in next iteration
    }

    if (tid == 0) {
        int c = 0;
        for (int i = 0; i < NROWS; ++i)
            if (cams[i] == cam) ++c;
        atomicAdd(out, total / (float)c);
    }
}

// ---------------------------------------------------------------------------
// Host launcher
// ---------------------------------------------------------------------------
extern "C" void kernel_launch(void* const* d_in, const int* in_sizes, int n_in,
                              void* d_out, int out_size, void* d_ws, size_t ws_size,
                              hipStream_t stream) {
    (void)in_sizes; (void)n_in; (void)out_size; (void)ws_size;

    const float* features = (const float*)d_in[0];
    // d_in[1] = global_features (unused by the reference computation)
    const float* memory   = (const float*)d_in[2];
    const int*   cams     = (const int*)d_in[3];
    const int*   proxy    = (const int*)d_in[4];
    float*       out      = (float*)d_out;

    char*   ws    = (char*)d_ws;
    __bf16* fbf   = (__bf16*)ws;                                   //  4 MB bf16 features
    float*  sims0 = (float*)(ws + (size_t)(4u << 20));             // 32 MB
    float*  sims1 = (float*)(ws + (size_t)(4u << 20) + (size_t)(32u << 20)); // 32 MB

    zero_out_kernel<<<1, 1, 0, stream>>>(out);

    // 512*4096 / (256 threads * 4 elems) = 2048 blocks
    cvt_feat_kernel<<<2048, 256, 0, stream>>>(features, fbf);

    // 16384 cols / 64 per block = 256 blocks per half
    gemm_bf16_kernel<<<256, 256, 0, stream>>>(fbf, memory, sims0, 0);
    gemm_bf16_kernel<<<256, 256, 0, stream>>>(fbf, memory, sims1, 1);

    // dynamic LDS: row (64 KB) + reduction/broadcast scratch
    size_t shmem = (size_t)(PCOLS + 256 + 64 + 32) * 4 + (size_t)(256 + 64 + 32) * 4;
    loss_kernel<<<NROWS, 256, shmem, stream>>>(sims0, sims1, cams, proxy, out);
}
